// MultiTaskPowerGNN_2568390443010
// MI455X (gfx1250) — matrix operand order repaired and checked by reading, not compiled, not run on hardware
//
#include <hip/hip_runtime.h>

#define NN 100000
#define EE 3200000
#define GG 64
#define HH 64
#define EPSV 1e-5f

typedef __attribute__((ext_vector_type(16))) _Float16 v16h;
typedef __attribute__((ext_vector_type(8)))  float    v8f;
typedef __attribute__((ext_vector_type(4)))  unsigned int v4u;
typedef __attribute__((ext_vector_type(8)))  int v8i;
typedef __attribute__((ext_vector_type(4)))  int v4i;

// ---------------- utility ----------------
__global__ void k_zero(float* __restrict__ p, int n) {
  int i = blockIdx.x * blockDim.x + threadIdx.x;
  int stride = gridDim.x * blockDim.x;
  for (; i < n; i += stride) p[i] = 0.0f;
}

// deg[d] += 1 for each edge (in-degree)
__global__ void k_degree(const int* __restrict__ dst, float* __restrict__ deg) {
  int e = blockIdx.x * blockDim.x + threadIdx.x;
  if (e < EE) atomicAdd(&deg[dst[e]], 1.0f);
}

// dis[n] = rsqrt(deg[n] + 1)   (self-loop)
__global__ void k_dis(float* __restrict__ deg) {
  int n = blockIdx.x * blockDim.x + threadIdx.x;
  if (n < NN) deg[n] = rsqrtf(deg[n] + 1.0f);
}

// ---------------- layer 0 GEMM: [N,3] @ [3,64] ----------------
__global__ void k_gemm3(const float* __restrict__ x, const float* __restrict__ W0,
                        float* __restrict__ h) {
  int i = blockIdx.x * blockDim.x + threadIdx.x;
  if (i >= NN * HH) return;
  int n = i >> 6, c = i & 63;
  const float* xr = x + n * 3;
  h[i] = xr[0] * W0[c] + xr[1] * W0[64 + c] + xr[2] * W0[128 + c];
}

// ---------------- WMMA GEMM: [N,64] @ [64,64] ----------------
// W is staged into LDS via the Tensor Data Mover, then re-laid out into
// B-fragment order: frag(ks,nt) -> 32 lanes x 16 contiguous halfs.
__global__ __launch_bounds__(256) void k_gemm_wmma(const float* __restrict__ A,
                                                   const float* __restrict__ W,
                                                   float* __restrict__ O) {
  __shared__ __align__(16) float    Wraw[64 * 64];   // raw f32 W tile (16 KB)
  __shared__ __align__(32) _Float16 Wfrag[64 * 64];  // B fragments (8 KB)

#if __has_builtin(__builtin_amdgcn_tensor_load_to_lds)
  // --- TDM: DMA the 64x64 f32 W tile into LDS ---
  if (threadIdx.x < 32) {                       // one wave issues; TDM ignores EXEC
    unsigned long long ga = (unsigned long long)(uintptr_t)W;
    unsigned int ldsoff = (unsigned int)(uintptr_t)(void*)Wraw;  // low 32 = LDS offset
    v4u g0;
    g0[0] = 1u;                                          // count=1 (valid), user mode
    g0[1] = ldsoff;                                      // lds_addr (bytes)
    g0[2] = (unsigned int)(ga & 0xFFFFFFFFu);            // global_addr[31:0]
    g0[3] = (unsigned int)((ga >> 32) & 0x1FFFFFFu)      // global_addr[56:32]
          | (2u << 30);                                  // type=2 (image)
    v8i g1;
    g1[0] = (int)(2u << 16);       // data_size=2 -> 4 bytes; no multicast/pad/iterate
    g1[1] = (int)(64u << 16);      // tensor_dim0[15:0]=64 in bits [63:48]
    g1[2] = (int)(64u << 16);      // tensor_dim0 hi=0 | tensor_dim1[15:0]=64
    g1[3] = (int)(64u << 16);      // tensor_dim1 hi=0 | tile_dim0=64
    g1[4] = 64;                    // tile_dim1=64, tile_dim2=0
    g1[5] = 64;                    // tensor_dim0_stride=64 (low 32)
    g1[6] = 0;
    g1[7] = 0;
    v4i gz  = {0, 0, 0, 0};
    v8i gz8 = {0, 0, 0, 0, 0, 0, 0, 0};
    __builtin_amdgcn_tensor_load_to_lds(g0, g1, gz, gz, gz8, 0);
    __builtin_amdgcn_s_wait_tensorcnt(0);
  }
  __syncthreads();
  const float* wsrc = Wraw;
#else
  const float* wsrc = W;           // fallback: convert straight from global
#endif

  // Re-layout into WMMA B-fragment order (16-bit B 32x16, wave32):
  // N=(lane&15)+nt*16; j<8: K=ks*32+hi*8+j ; j>=8: K=ks*32+16+hi*8+(j-8)
  for (int i = threadIdx.x; i < 64 * 64; i += 256) {
    int k = i >> 6, n = i & 63;
    int ks = k >> 5, kk = k & 31;
    int hi2 = (kk >> 3) & 1, j = (kk & 7) | ((kk >> 4) << 3);
    int nt = n >> 4;
    int ln = (n & 15) + hi2 * 16;
    Wfrag[((((ks << 2) | nt) * 32) + ln) * 16 + j] = (_Float16)wsrc[i];
  }
  __syncthreads();

  int wv   = (blockIdx.x * 256 + threadIdx.x) >> 5;  // global wave id
  int lane = threadIdx.x & 31;
  int row0 = wv * 16;
  if (row0 < NN) {                 // uniform per wave -> EXEC all ones inside
    int m  = lane & 15;
    int hi = lane >> 4;            // 0 or 1
    const v16h* BF = (const v16h*)Wfrag;
    v8f acc0 = {}, acc1 = {}, acc2 = {}, acc3 = {};
#pragma unroll
    for (int ks = 0; ks < 2; ++ks) {
      // A fragment (16-bit A 16x32 layout): lane holds K {b..b+7, b+16..b+23},
      // b = ks*32 + hi*8
      const float4* ar = (const float4*)(A + (size_t)(row0 + m) * 64 + ks * 32 + hi * 8);
      float4 p0 = ar[0], p1 = ar[1];   // K b..b+7
      float4 p2 = ar[4], p3 = ar[5];   // K b+16..b+23
      v16h a;
      a[0]  = (_Float16)p0.x; a[1]  = (_Float16)p0.y;
      a[2]  = (_Float16)p0.z; a[3]  = (_Float16)p0.w;
      a[4]  = (_Float16)p1.x; a[5]  = (_Float16)p1.y;
      a[6]  = (_Float16)p1.z; a[7]  = (_Float16)p1.w;
      a[8]  = (_Float16)p2.x; a[9]  = (_Float16)p2.y;
      a[10] = (_Float16)p2.z; a[11] = (_Float16)p2.w;
      a[12] = (_Float16)p3.x; a[13] = (_Float16)p3.y;
      a[14] = (_Float16)p3.z; a[15] = (_Float16)p3.w;

      int fb = (ks << 2) * 32 + lane;
      acc0 = __builtin_amdgcn_wmma_f32_16x16x32_f16(false, a, false, BF[fb + 0 * 32],
                                                    (short)0, acc0, false, false);
      acc1 = __builtin_amdgcn_wmma_f32_16x16x32_f16(false, a, false, BF[fb + 1 * 32],
                                                    (short)0, acc1, false, false);
      acc2 = __builtin_amdgcn_wmma_f32_16x16x32_f16(false, a, false, BF[fb + 2 * 32],
                                                    (short)0, acc2, false, false);
      acc3 = __builtin_amdgcn_wmma_f32_16x16x32_f16(false, a, false, BF[fb + 3 * 32],
                                                    (short)0, acc3, false, false);
    }
    // C/D layout: VGPR r, lane<16 -> (M=r, N=m), lane>=16 -> (M=r+8, N=m)
#pragma unroll
    for (int r = 0; r < 8; ++r) {
      size_t orow = (size_t)(row0 + r + 8 * hi) * 64 + m;
      O[orow + 0]  = acc0[r];
      O[orow + 16] = acc1[r];
      O[orow + 32] = acc2[r];
      O[orow + 48] = acc3[r];
    }
  }
}

// ---------------- edge scatter: agg[dst] += h[src]*dis[src]*dis[dst] ----------------
// 16 threads x float4 per edge -> coalesced b128 gather + contiguous atomic burst
__global__ void k_edge(const int* __restrict__ ei, const float* __restrict__ h,
                       const float* __restrict__ dis, float* __restrict__ agg) {
  int gid = blockIdx.x * blockDim.x + threadIdx.x;   // < E*16 = 51.2M
  int e = gid >> 4, q = gid & 15;
  int s = ei[e];
  int d = ei[EE + e];
  float coef = dis[s] * dis[d];
  float4 hv = reinterpret_cast<const float4*>(h)[s * 16 + q];
  float* ap = agg + (size_t)d * 64 + q * 4;
  atomicAdd(ap + 0, hv.x * coef);
  atomicAdd(ap + 1, hv.y * coef);
  atomicAdd(ap + 2, hv.z * coef);
  atomicAdd(ap + 3, hv.w * coef);
}

// ---------------- y = agg + h*dis^2 + b (in place) + BN partial stats ----------------
__global__ void k_post(float* __restrict__ agg, const float* __restrict__ h,
                       const float* __restrict__ dis, const float* __restrict__ b,
                       float* __restrict__ sums) {   // sums[0:64]=sum, [64:128]=sumsq
  int tid = blockIdx.x * blockDim.x + threadIdx.x;   // 65536 threads
  int c = tid & 63;
  int stride = (gridDim.x * blockDim.x) >> 6;
  float bc = b[c], s = 0.0f, s2 = 0.0f;
  for (int n = tid >> 6; n < NN; n += stride) {
    size_t idx = (size_t)n * 64 + c;
    float dd = dis[n];
    float y = agg[idx] + h[idx] * (dd * dd) + bc;
    agg[idx] = y;
    s += y;
    s2 += y * y;
  }
  atomicAdd(&sums[c], s);
  atomicAdd(&sums[64 + c], s2);
}

__global__ void k_bnfin(const float* __restrict__ sums, const float* __restrict__ g,
                        const float* __restrict__ be, float* __restrict__ sc) {
  int c = threadIdx.x;
  if (c < 64) {
    float m = sums[c] * (1.0f / NN);
    float v = sums[64 + c] * (1.0f / NN) - m * m;
    float scale = g[c] * rsqrtf(v + EPSV);
    sc[c] = scale;
    sc[64 + c] = be[c] - m * scale;
  }
}

__global__ void k_bnapply(const float* __restrict__ agg, const float* __restrict__ sc,
                          float* __restrict__ act) {
  int i = blockIdx.x * blockDim.x + threadIdx.x;
  if (i < NN * HH) {
    int c = i & 63;
    act[i] = fmaxf(sc[c] * agg[i] + sc[64 + c], 0.0f);
  }
}

// ---------------- pooling ----------------
__global__ void k_pool(const float* __restrict__ act, const int* __restrict__ batch,
                       float* __restrict__ pool, float* __restrict__ cnt) {
  int i = blockIdx.x * blockDim.x + threadIdx.x;
  if (i < NN * HH) {
    int n = i >> 6, c = i & 63;
    int g = batch[n];
    atomicAdd(&pool[g * 64 + c], act[i]);
    if (c == 0) atomicAdd(&cnt[g], 1.0f);
  }
}

// ---------------- heads ----------------
__global__ void k_heads(float* __restrict__ pool, const float* __restrict__ cnt,
                        const float* pW1, const float* pb1, const float* pW2, const float* pb2,
                        const float* aW1, const float* ab1, const float* aW2, const float* ab2,
                        const float* tW1, const float* tb1, const float* tW2, const float* tb2,
                        float* __restrict__ out) {
  int tid = threadIdx.x;
  for (int i = tid; i < GG * HH; i += blockDim.x)
    pool[i] = pool[i] / fmaxf(cnt[i >> 6], 1.0f);
  __syncthreads();
  if (tid < GG) {
    const float* pr = pool + tid * 64;
    auto head1 = [&](const float* W1h, const float* b1h, const float* W2h,
                     const float* b2h) -> float {
      float o = b2h[0];
      for (int j = 0; j < 32; ++j) {
        float s = b1h[j];
        for (int c = 0; c < 64; ++c) s += pr[c] * W1h[c * 32 + j];
        o += fmaxf(s, 0.0f) * W2h[j];
      }
      return o;
    };
    out[tid * 4 + 0] = head1(pW1, pb1, pW2, pb2);
    out[tid * 4 + 1] = head1(aW1, ab1, aW2, ab2);
    float t0 = tb2[0], t1 = tb2[1];
    for (int j = 0; j < 32; ++j) {
      float s = tb1[j];
      for (int c = 0; c < 64; ++c) s += pr[c] * tW1[c * 32 + j];
      s = fmaxf(s, 0.0f);
      t0 += s * tW2[j * 2 + 0];
      t1 += s * tW2[j * 2 + 1];
    }
    out[tid * 4 + 2] = t0;
    out[tid * 4 + 3] = t1;
  }
}

// ---------------- launcher ----------------
extern "C" void kernel_launch(void* const* d_in, const int* in_sizes, int n_in,
                              void* d_out, int out_size, void* d_ws, size_t ws_size,
                              hipStream_t stream) {
  const float* x     = (const float*)d_in[0];
  const int*   ei    = (const int*)d_in[1];
  const int*   batch = (const int*)d_in[2];
  const float* W[3]  = {(const float*)d_in[3], (const float*)d_in[7], (const float*)d_in[11]};
  const float* bi[3] = {(const float*)d_in[4], (const float*)d_in[8], (const float*)d_in[12]};
  const float* gm[3] = {(const float*)d_in[5], (const float*)d_in[9], (const float*)d_in[13]};
  const float* bt[3] = {(const float*)d_in[6], (const float*)d_in[10], (const float*)d_in[14]};

  float* ws   = (float*)d_ws;
  float* deg  = ws;                       // N floats (becomes dis)
  float* bufA = ws + 100000;              // N*64
  float* bufB = bufA + (size_t)NN * 64;   // N*64
  float* agg  = bufB + (size_t)NN * 64;   // N*64
  float* sums = agg + (size_t)NN * 64;    // 128
  float* sc   = sums + 128;               // 128
  float* pool = sc + 128;                 // 4096
  float* cnt  = pool + 4096;              // 64

  dim3 B(256);

  // degree -> dis
  k_zero<<<256, B, 0, stream>>>(deg, NN);
  k_degree<<<(EE + 255) / 256, B, 0, stream>>>(ei + EE, deg);
  k_dis<<<(NN + 255) / 256, B, 0, stream>>>(deg);

  for (int l = 0; l < 3; ++l) {
    float* h = bufB;
    if (l == 0)
      k_gemm3<<<(NN * HH + 255) / 256, B, 0, stream>>>(x, W[0], h);
    else
      k_gemm_wmma<<<(NN / 16 + 7) / 8, B, 0, stream>>>(bufA, W[l], h);
    k_zero<<<4096, B, 0, stream>>>(agg, NN * HH);
    k_edge<<<EE * 16 / 256, B, 0, stream>>>(ei, h, deg, agg);
    k_zero<<<1, 128, 0, stream>>>(sums, 128);
    k_post<<<256, B, 0, stream>>>(agg, h, deg, bi[l], sums);
    k_bnfin<<<1, 64, 0, stream>>>(sums, gm[l], bt[l], sc);
    k_bnapply<<<(NN * HH + 255) / 256, B, 0, stream>>>(agg, sc, bufA);
  }

  k_zero<<<17, B, 0, stream>>>(pool, 4096 + 64);
  k_pool<<<(NN * HH + 255) / 256, B, 0, stream>>>(bufA, batch, pool, cnt);
  k_heads<<<1, B, 0, stream>>>(pool, cnt,
      (const float*)d_in[15], (const float*)d_in[16], (const float*)d_in[17], (const float*)d_in[18],
      (const float*)d_in[19], (const float*)d_in[20], (const float*)d_in[21], (const float*)d_in[22],
      (const float*)d_in[23], (const float*)d_in[24], (const float*)d_in[25], (const float*)d_in[26],
      (float*)d_out);
}
